// Attention_18708877541532
// MI455X (gfx1250) — matrix operand order repaired and checked by reading, not compiled, not run on hardware
//
#include <hip/hip_runtime.h>

// ---------------------------------------------------------------------------
// MI455X (gfx1250) fused cross-attention.
//   - bf16 WMMA (v_wmma_f32_16x16x32_bf16) for all GEMMs
//   - weights pre-transposed so all A/B fragments are contiguous 16B loads
//   - flash-style fused attention (no scores/attn materialized in HBM)
//   - K/V and GEMM-B tiles staged in LDS once per block (kills 8x redundant
//     L2 traffic across the 8 waves) via GLOBAL_LOAD_ASYNC_TO_LDS_B128
//     (ASYNCcnt + s_wait_asynccnt), falling back to vector copies if absent
//   - wave32 layouts per CDNA5 ISA 7.12.2
// ---------------------------------------------------------------------------

static constexpr int Bx    = 16;
static constexpr int QL    = 1024;
static constexpr int KL    = 512;
static constexpr int EMB   = 768;
static constexpr int HEADS = 12;
static constexpr int DH    = 64;
static constexpr int INNER = 768;

typedef __attribute__((ext_vector_type(16))) __bf16       v16bf;
typedef __attribute__((ext_vector_type(8)))  float        v8f;
typedef __attribute__((ext_vector_type(4)))  unsigned int v4u;
typedef __attribute__((ext_vector_type(4)))  int          v4i;

union FragBF16 {
  v16bf v;
  v4u   u[2];
};

#if __has_builtin(__builtin_amdgcn_global_load_async_to_lds_b128)
#define USE_ASYNC_LDS 1
#else
#define USE_ASYNC_LDS 0
#endif

// per-lane 16B global -> LDS stage (CDNA5 GLOBAL_LOAD_ASYNC_TO_LDS_B128 when
// available; tracked by ASYNCcnt), else a plain register-path copy.
__device__ __forceinline__ void stage_b128(void* lds_dst, const void* gsrc) {
#if USE_ASYNC_LDS
  __builtin_amdgcn_global_load_async_to_lds_b128(
      (__attribute__((address_space(1))) v4i*)(unsigned long long)gsrc,
      (__attribute__((address_space(3))) v4i*)(unsigned)(unsigned long long)lds_dst,
      0, 0);
#else
  *reinterpret_cast<v4u*>(lds_dst) = *reinterpret_cast<const v4u*>(gsrc);
#endif
}

__device__ __forceinline__ void stage_wait_barrier() {
#if USE_ASYNC_LDS
#if __has_builtin(__builtin_amdgcn_s_wait_asynccnt)
  __builtin_amdgcn_s_wait_asynccnt(0);
#else
  asm volatile("s_wait_asynccnt 0x0" ::: "memory");
#endif
#endif
  __syncthreads();
}

__device__ __forceinline__ unsigned short f2bf(float f) {
  unsigned u = __float_as_uint(f);
  unsigned r = (u + 0x7FFFu + ((u >> 16) & 1u)) >> 16;  // round-to-nearest-even
  return (unsigned short)r;
}

__device__ __forceinline__ v8f wmma_bf16(const FragBF16& a, const FragBF16& b, v8f c) {
  // D = A(16x32 bf16) * B(32x16 bf16) + C(16x16 f32)
  return __builtin_amdgcn_wmma_f32_16x16x32_bf16(false, a.v, false, b.v,
                                                 (short)0, c, false, false);
}

__device__ __forceinline__ v8f vsplat8(float x) {
  v8f r;
#pragma unroll
  for (int i = 0; i < 8; i++) r[i] = x;
  return r;
}

__device__ __forceinline__ v8f vmax8(v8f a, v8f b) {
  v8f r;
#pragma unroll
  for (int i = 0; i < 8; i++) r[i] = fmaxf(a[i], b[i]);
  return r;
}

// reductions across the 16 lanes that hold the same rows (wave32 halves)
__device__ __forceinline__ v8f rowmax_half16(v8f x) {
#pragma unroll
  for (int off = 8; off >= 1; off >>= 1) {
#pragma unroll
    for (int c = 0; c < 8; c++) x[c] = fmaxf(x[c], __shfl_xor(x[c], off, 32));
  }
  return x;
}

__device__ __forceinline__ v8f rowsum_half16(v8f x) {
#pragma unroll
  for (int off = 8; off >= 1; off >>= 1) {
#pragma unroll
    for (int c = 0; c < 8; c++) x[c] += __shfl_xor(x[c], off, 32);
  }
  return x;
}

// ---------------------------------------------------------------------------
// fp32 -> bf16 conversion (grid-stride)
// ---------------------------------------------------------------------------
__global__ void k_to_bf16(const float* __restrict__ src,
                          unsigned short* __restrict__ dst, int n) {
  int i = blockIdx.x * blockDim.x + threadIdx.x;
  int stride = gridDim.x * blockDim.x;
  for (; i < n; i += stride) dst[i] = f2bf(src[i]);
}

// ---------------------------------------------------------------------------
// Weight transpose + bf16 convert: Wt[c*R + r] = bf16(W[r*C + c])
// ---------------------------------------------------------------------------
__global__ void k_transpose_bf16(const float* __restrict__ W,
                                 unsigned short* __restrict__ Wt, int R, int C) {
  __shared__ float tile[32][33];
  const int bx = blockIdx.x * 32;  // source col block
  const int by = blockIdx.y * 32;  // source row block
  const int x = bx + threadIdx.x;
#pragma unroll
  for (int i = threadIdx.y; i < 32; i += 8)
    tile[i][threadIdx.x] = W[(size_t)(by + i) * C + x];
  __syncthreads();
  const int xo = by + threadIdx.x;  // output inner index = source row
#pragma unroll
  for (int i = threadIdx.y; i < 32; i += 8)
    Wt[(size_t)(bx + i) * R + xo] = f2bf(tile[threadIdx.x][i]);
}

// ---------------------------------------------------------------------------
// Per-key bias: tanh(fus_w0*table[idx] + fus_w1*(val*vp_w+vp_b) + fus_b),
// clipped to [-5,5]; masked keys get -1e9 (dominates softmax).
// ---------------------------------------------------------------------------
__global__ void k_bias(const float* __restrict__ lab_values,
                       const float* __restrict__ bias_table,
                       const float* __restrict__ vp_w, const float* __restrict__ vp_b,
                       const float* __restrict__ fus_w, const float* __restrict__ fus_b,
                       const int* __restrict__ idxs, const int* __restrict__ mask,
                       float* __restrict__ biasv, int n) {
  int i = blockIdx.x * blockDim.x + threadIdx.x;
  if (i >= n) return;
  float tb = bias_table[idxs[i]];
  float vb = lab_values[i] * vp_w[0] + vp_b[0];
  float t  = tanhf(tb * fus_w[0] + vb * fus_w[1] + fus_b[0]);
  t = fminf(5.0f, fmaxf(-5.0f, t));
  biasv[i] = mask[i] ? t : -1e9f;
}

// ---------------------------------------------------------------------------
// bf16 WMMA GEMM: C[M,N] = A[M,K] * Bt[N,K]^T
//   block = 256 threads (8 waves); each wave -> 16 rows x 64 cols.
//   The 64x32 B tile (shared by all 8 waves) is staged in LDS per K step.
//   OUTMODE 0: bf16 row-major
//   OUTMODE 1: bf16 per-head transpose -> vT[b][h][d][key]   (M=Bx*KL, N=INNER)
//   OUTMODE 2: f32 row-major with +bo[n] and +resid          (final output)
// ---------------------------------------------------------------------------
template <int OUTMODE>
__global__ __launch_bounds__(256) void k_gemm_bf16(
    const unsigned short* __restrict__ A, const unsigned short* __restrict__ Bt,
    void* __restrict__ Cout, const float* __restrict__ bo,
    const float* __restrict__ resid, int M, int N, int K) {
  const int wave = threadIdx.x >> 5;
  const int lane = threadIdx.x & 31;
  const int hf   = lane >> 4;   // which 16-lane half
  const int ln   = lane & 15;
  const int mBase = blockIdx.y * 128 + wave * 16;
  const int nBase = blockIdx.x * 64;

  // 64 B-rows x 32 K, padded to 40 elems (80B row stride: 16B-aligned chunks,
  // 20-dword bank stride -> spread over the 64 banks)
  __shared__ __align__(16) unsigned short ldsB[64][40];

  v8f acc[4];
#pragma unroll
  for (int j = 0; j < 4; j++) acc[j] = vsplat8(0.0f);

  const unsigned short* arow = A + (size_t)(mBase + ln) * K;
  const int fr  = threadIdx.x >> 2;        // fill row 0..63
  const int fc8 = (threadIdx.x & 3) * 8;   // fill col chunk

  for (int kk = 0; kk < K; kk += 32) {
    // cooperative stage of B tile: 64 rows x 32 elems = 256 x 16B chunks
    stage_b128(&ldsB[fr][fc8], Bt + (size_t)(nBase + fr) * K + kk + fc8);

    // A fragment (16x32): lanes 0-15 K kk..kk+7 / kk+16..23; lanes 16-31 +8
    FragBF16 fa;
    const unsigned short* ap = arow + kk + hf * 8;
    fa.u[0] = *reinterpret_cast<const v4u*>(ap);
    fa.u[1] = *reinterpret_cast<const v4u*>(ap + 16);
    __builtin_prefetch(ap + 64, 0, 0);  // global_prefetch K tile two steps ahead

    stage_wait_barrier();
#pragma unroll
    for (int j = 0; j < 4; j++) {
      // B fragment (32x16): lane col n, lanes 0-15 K kk..kk+7 / kk+16..23; lanes 16-31 +8
      FragBF16 fb;
      const unsigned short* bp = &ldsB[j * 16 + ln][hf * 8];
      fb.u[0] = *reinterpret_cast<const v4u*>(bp);
      fb.u[1] = *reinterpret_cast<const v4u*>(bp + 16);
      acc[j] = wmma_bf16(fa, fb, acc[j]);
    }
    __syncthreads();  // tile consumed; safe to refill
  }

  // C layout: lane col n = nBase+j*16+ln ; VGPR v -> row mBase + v + hf*8
#pragma unroll
  for (int j = 0; j < 4; j++) {
    const int n = nBase + j * 16 + ln;
#pragma unroll
    for (int v = 0; v < 8; v++) {
      const int m = mBase + v + hf * 8;
      if constexpr (OUTMODE == 0) {
        ((unsigned short*)Cout)[(size_t)m * N + n] = f2bf(acc[j][v]);
      } else if constexpr (OUTMODE == 1) {
        const int b = m >> 9, kpos = m & 511;     // M rows are (b, key)
        const int h = n >> 6, d = n & 63;         // N cols are (h, d)
        ((unsigned short*)Cout)[(((size_t)(b * HEADS + h) * DH + d) * KL) + kpos] =
            f2bf(acc[j][v]);
      } else {
        ((float*)Cout)[(size_t)m * N + n] =
            acc[j][v] + bo[n] + resid[(size_t)m * N + n];
      }
    }
  }
}

// ---------------------------------------------------------------------------
// Fused flash attention per (b,h):  att = softmax(q k^T * scale + bias) v
//   grid = (B*H, QL/128), block = 256 (8 waves); each wave owns 16 q-rows and
//   streams all 512 keys in blocks of 64 with online softmax.
//   K tile (64 keys x 64 d) and V tile (64 d x 64 keys, from vT) are staged
//   in LDS once per key block and shared by all 8 waves.
// ---------------------------------------------------------------------------
__global__ __launch_bounds__(256) void k_flash_attn(
    const unsigned short* __restrict__ qbf, const unsigned short* __restrict__ kbf,
    const unsigned short* __restrict__ vt, const float* __restrict__ biasv,
    unsigned short* __restrict__ att) {
  const int bh = blockIdx.x;          // b*HEADS + h
  const int b  = bh / HEADS;
  const int h  = bh % HEADS;
  const int qt = blockIdx.y;
  const int wave = threadIdx.x >> 5;
  const int lane = threadIdx.x & 31;
  const int hf   = lane >> 4;
  const int ln   = lane & 15;
  const int qBase = qt * 128 + wave * 16;
  const float scale = 0.125f;  // 64^-0.5

  // wave-private staging tile for P (C-layout -> A-layout transpose)
  __shared__ __align__(16) unsigned short ldsP[8][16][72];
  // block-shared K / V tiles (rows padded to 72 elems = 144B, 16B aligned)
  __shared__ __align__(16) unsigned short ldsK[64][72];  // [key][d]
  __shared__ __align__(16) unsigned short ldsV[64][72];  // [d][key]

  // Q A-fragments, resident for whole kernel: 16 rows x 64 d = 2 K-steps
  FragBF16 fq[2];
#pragma unroll
  for (int ks = 0; ks < 2; ks++) {
    const unsigned short* qp =
        qbf + (size_t)(b * QL + qBase + ln) * INNER + h * DH + ks * 32 + hf * 8;
    fq[ks].u[0] = *reinterpret_cast<const v4u*>(qp);
    fq[ks].u[1] = *reinterpret_cast<const v4u*>(qp + 16);
  }

  v8f o[4];
#pragma unroll
  for (int j = 0; j < 4; j++) o[j] = vsplat8(0.0f);
  v8f mrow = vsplat8(-3.0e38f);
  v8f lrow = vsplat8(0.0f);

  const unsigned short* kblk = kbf + (size_t)(b * KL) * INNER + h * DH;
  const unsigned short* vblk = vt + (size_t)(bh * DH) * KL;

  for (int kb = 0; kb < KL / 64; kb++) {
    // ---- cooperative stage of K and V tiles: 2 x 512 chunks of 16B ----
#pragma unroll
    for (int c = threadIdx.x; c < 512; c += 256) {
      const int r = c >> 3, c8 = (c & 7) * 8;
      stage_b128(&ldsK[r][c8], kblk + (size_t)(kb * 64 + r) * INNER + c8);
      stage_b128(&ldsV[r][c8], vblk + (size_t)r * KL + kb * 64 + c8);
    }
    stage_wait_barrier();

    // ---- S = Q K^T for a 16x64 strip ----
    v8f s[4];
#pragma unroll
    for (int j = 0; j < 4; j++) s[j] = vsplat8(0.0f);
#pragma unroll
    for (int j = 0; j < 4; j++) {
#pragma unroll
      for (int ks = 0; ks < 2; ks++) {
        FragBF16 fk;  // B fragment: col = key, K-dim = d (contiguous in ldsK)
        const unsigned short* kp = &ldsK[j * 16 + ln][ks * 32 + hf * 8];
        fk.u[0] = *reinterpret_cast<const v4u*>(kp);
        fk.u[1] = *reinterpret_cast<const v4u*>(kp + 16);
        s[j] = wmma_bf16(fq[ks], fk, s[j]);
      }
    }

    // ---- scale + per-key bias (mask folded as -1e9) ----
#pragma unroll
    for (int j = 0; j < 4; j++) {
      const int key = kb * 64 + j * 16 + ln;
      const float bv = biasv[b * KL + key];
      s[j] = s[j] * scale + bv;
    }

    // ---- online softmax (row stats live in C-layout component form) ----
    v8f mx = vmax8(vmax8(s[0], s[1]), vmax8(s[2], s[3]));
    mx = rowmax_half16(mx);
    v8f mnew = vmax8(mrow, mx);
    v8f alpha;
#pragma unroll
    for (int c = 0; c < 8; c++) alpha[c] = __expf(mrow[c] - mnew[c]);
#pragma unroll
    for (int j = 0; j < 4; j++)
#pragma unroll
      for (int c = 0; c < 8; c++) s[j][c] = __expf(s[j][c] - mnew[c]);
    v8f rs = rowsum_half16(s[0] + s[1] + s[2] + s[3]);
    lrow = lrow * alpha + rs;
    mrow = mnew;
#pragma unroll
    for (int j = 0; j < 4; j++) o[j] = o[j] * alpha;

    // ---- P: C-layout -> LDS -> A-layout fragments (wave-private tile) ----
#pragma unroll
    for (int j = 0; j < 4; j++)
#pragma unroll
      for (int v = 0; v < 8; v++)
        ldsP[wave][v + hf * 8][j * 16 + ln] = f2bf(s[j][v]);
    asm volatile("s_wait_dscnt 0x0" ::: "memory");  // CDNA5 split DS counter

    FragBF16 fp[2];
#pragma unroll
    for (int ks = 0; ks < 2; ks++) {
      const unsigned short* pp = &ldsP[wave][ln][ks * 32 + hf * 8];
      fp[ks].u[0] = *reinterpret_cast<const v4u*>(pp);
      fp[ks].u[1] = *reinterpret_cast<const v4u*>(pp + 16);
    }

    // ---- O += P V : B fragments from staged vT (key contiguous per d) ----
#pragma unroll
    for (int j = 0; j < 4; j++) {
#pragma unroll
      for (int ks = 0; ks < 2; ks++) {
        FragBF16 fv;
        const unsigned short* vp = &ldsV[j * 16 + ln][ks * 32 + hf * 8];
        fv.u[0] = *reinterpret_cast<const v4u*>(vp);
        fv.u[1] = *reinterpret_cast<const v4u*>(vp + 16);
        o[j] = wmma_bf16(fp[ks], fv, o[j]);
      }
    }
    __syncthreads();  // K/V tiles consumed; safe to refill
  }

  // ---- normalize and store attended (bf16 row-major [B*QL, INNER]) ----
#pragma unroll
  for (int j = 0; j < 4; j++) {
    const int n = h * DH + j * 16 + ln;
#pragma unroll
    for (int v = 0; v < 8; v++) {
      const int row = b * QL + qBase + v + hf * 8;
      att[(size_t)row * INNER + n] = f2bf(o[j][v] / lrow[v]);
    }
  }
}

// ---------------------------------------------------------------------------
// Host-side orchestration
// ---------------------------------------------------------------------------
extern "C" void kernel_launch(void* const* d_in, const int* in_sizes, int n_in,
                              void* d_out, int out_size, void* d_ws, size_t ws_size,
                              hipStream_t stream) {
  (void)in_sizes; (void)n_in; (void)out_size; (void)ws_size;

  const float* imgE  = (const float*)d_in[0];
  const float* labE  = (const float*)d_in[1];
  const float* labV  = (const float*)d_in[2];
  const float* Wq    = (const float*)d_in[3];
  const float* Wk    = (const float*)d_in[4];
  const float* Wv    = (const float*)d_in[5];
  const float* Wo    = (const float*)d_in[6];
  const float* bo    = (const float*)d_in[7];
  const float* btab  = (const float*)d_in[8];
  const float* vp_w  = (const float*)d_in[9];
  const float* vp_b  = (const float*)d_in[10];
  const float* fus_w = (const float*)d_in[11];
  const float* fus_b = (const float*)d_in[12];
  const int*   idxs  = (const int*)d_in[13];
  const int*   mask  = (const int*)d_in[14];

  char* ws = (char*)d_ws;
  size_t off = 0;
  auto take = [&](size_t bytes) -> char* {
    char* p = ws + off;
    off = (off + bytes + 255) & ~(size_t)255;
    return p;
  };
  const size_t nImg = (size_t)Bx * QL * EMB;   // 12.58M
  const size_t nLab = (size_t)Bx * KL * EMB;   //  6.29M
  const size_t nW   = (size_t)EMB * INNER;

  unsigned short* imgE_bf = (unsigned short*)take(nImg * 2);
  unsigned short* labE_bf = (unsigned short*)take(nLab * 2);
  unsigned short* WqT     = (unsigned short*)take(nW * 2);
  unsigned short* WkT     = (unsigned short*)take(nW * 2);
  unsigned short* WvT     = (unsigned short*)take(nW * 2);
  unsigned short* WoT     = (unsigned short*)take(nW * 2);
  unsigned short* qbf     = (unsigned short*)take(nImg * 2);
  unsigned short* kbf     = (unsigned short*)take(nLab * 2);
  unsigned short* vtbf    = (unsigned short*)take(nLab * 2);
  unsigned short* attb    = (unsigned short*)take(nImg * 2);
  float*          biasv   = (float*)take((size_t)Bx * KL * 4);

  // 1) fp32 -> bf16 staging
  k_to_bf16<<<2048, 256, 0, stream>>>(imgE, imgE_bf, (int)nImg);
  k_to_bf16<<<2048, 256, 0, stream>>>(labE, labE_bf, (int)nLab);

  // 2) weight transposes (B fragments need Bt[N][K])
  dim3 tb(32, 8), tg(EMB / 32, EMB / 32);
  k_transpose_bf16<<<tg, tb, 0, stream>>>(Wq, WqT, EMB, INNER);
  k_transpose_bf16<<<tg, tb, 0, stream>>>(Wk, WkT, EMB, INNER);
  k_transpose_bf16<<<tg, tb, 0, stream>>>(Wv, WvT, EMB, INNER);
  k_transpose_bf16<<<tg, tb, 0, stream>>>(Wo, WoT, INNER, EMB);

  // 3) per-key bias (+mask folded)
  k_bias<<<(Bx * KL + 255) / 256, 256, 0, stream>>>(
      labV, btab, vp_w, vp_b, fus_w, fus_b, idxs, mask, biasv, Bx * KL);

  // 4) projections
  k_gemm_bf16<0><<<dim3(INNER / 64, (Bx * QL) / 128), 256, 0, stream>>>(
      imgE_bf, WqT, qbf, nullptr, nullptr, Bx * QL, INNER, EMB);
  k_gemm_bf16<0><<<dim3(INNER / 64, (Bx * KL) / 128), 256, 0, stream>>>(
      labE_bf, WkT, kbf, nullptr, nullptr, Bx * KL, INNER, EMB);
  k_gemm_bf16<1><<<dim3(INNER / 64, (Bx * KL) / 128), 256, 0, stream>>>(
      labE_bf, WvT, vtbf, nullptr, nullptr, Bx * KL, INNER, EMB);

  // 5) fused attention (no scores/attn in HBM)
  k_flash_attn<<<dim3(Bx * HEADS, QL / 128), 256, 0, stream>>>(
      qbf, kbf, vtbf, biasv, attb);

  // 6) output projection + bias + residual (f32 out)
  k_gemm_bf16<2><<<dim3(EMB / 64, (Bx * QL) / 128), 256, 0, stream>>>(
      attb, WoT, d_out, bo, imgE, Bx * QL, EMB, INNER);
}
